// MoeBlock_35175782154270
// MI455X (gfx1250) — compile-verified, matrix operand order checked by reading, not run on hardware
//
#include <hip/hip_runtime.h>

typedef __bf16 bf16_t;
typedef __attribute__((ext_vector_type(16))) __bf16 v16bf;
typedef __attribute__((ext_vector_type(8)))  __bf16 bf16x8;
typedef __attribute__((ext_vector_type(8)))  float  v8f;

#define T_TOK 16384
#define DIM 512
#define NEXP 8
#define TILE_M 32
#define MAX_TILES 1032            // ceil((2T + 8*31)/32) rounded up
#define P_CAP (MAX_TILES * TILE_M)
#define LDS_STRIDE 520            // 512 + 8 bf16 pad -> rows spread across banks

// ---------------------------------------------------------------------------
// 1) Router: NCHW -> [T,512] bf16 tokens, logits, top-2 renormalized weights.
// ---------------------------------------------------------------------------
__global__ void k_router(const float* __restrict__ hs, const float* __restrict__ gate_w,
                         bf16_t* __restrict__ tok, int* __restrict__ routee,
                         float* __restrict__ routew) {
  int bh = blockIdx.x;                 // 0..255
  int b = bh >> 6, h = bh & 63;
  int tx = threadIdx.x & 63;           // w
  int ty = threadIdx.x >> 6;           // 0..3
  int t = (b << 12) + (h << 6) + tx;   // token id
  const float* base = hs + (size_t)b * DIM * 4096 + (h << 6) + tx;  // hs[b][c][h][w]

  float acc[NEXP];
#pragma unroll
  for (int e = 0; e < NEXP; e++) acc[e] = 0.f;

  for (int c = ty; c < DIM; c += 4) {
    float v = base[(size_t)c * 4096];
    tok[(size_t)t * DIM + c] = (bf16_t)v;
    const float* g = gate_w + c * NEXP;
#pragma unroll
    for (int e = 0; e < NEXP; e++) acc[e] += v * g[e];
  }

  __shared__ float red[4][64][NEXP];
#pragma unroll
  for (int e = 0; e < NEXP; e++) red[ty][tx][e] = acc[e];
  __syncthreads();

  if (ty == 0) {
    float lg[NEXP];
#pragma unroll
    for (int e = 0; e < NEXP; e++)
      lg[e] = red[0][tx][e] + red[1][tx][e] + red[2][tx][e] + red[3][tx][e];
    int i0 = 0;
    for (int e = 1; e < NEXP; e++) if (lg[e] > lg[i0]) i0 = e;
    int i1 = (i0 == 0) ? 1 : 0;
    for (int e = 0; e < NEXP; e++) if (e != i0 && lg[e] > lg[i1]) i1 = e;
    // softmax denom cancels under top-2 renorm: w0 = 1/(1+exp(l1-l0))
    float p1 = __expf(lg[i1] - lg[i0]);
    float s = 1.f + p1;
    routee[2 * t] = i0;  routew[2 * t] = 1.f / s;
    routee[2 * t + 1] = i1;  routew[2 * t + 1] = p1 / s;
  }
}

// ---------------------------------------------------------------------------
// 2) Weight convert: fp32 [E][Din][Dout] -> bf16 [E][Dout][Din] (B col-major)
// ---------------------------------------------------------------------------
__global__ void k_convw(const float* __restrict__ w, bf16_t* __restrict__ wt) {
  size_t total = (size_t)NEXP * DIM * DIM;
  for (size_t idx = (size_t)blockIdx.x * 256 + threadIdx.x; idx < total;
       idx += (size_t)gridDim.x * 256) {
    size_t k = idx % DIM;
    size_t rest = idx / DIM;
    size_t n = rest % DIM;
    size_t e = rest / DIM;
    wt[idx] = (bf16_t)w[(e * DIM + k) * DIM + n];
  }
}

// ---------------------------------------------------------------------------
// 3a) Count: expert histogram, 32-padded bases, tile->expert map, zero row.
//     meta: [0..7]=cnt  [8..16]=padBase(9)  [17]=nTiles  [18..]=tile_expert
// ---------------------------------------------------------------------------
__global__ void k_count(const int* __restrict__ routee, int* __restrict__ meta,
                        bf16_t* __restrict__ tok) {
  __shared__ int hist[NEXP];
  __shared__ int pb[NEXP + 1];
  __shared__ int ntil;
  int tid = threadIdx.x;
  if (tid < NEXP) hist[tid] = 0;
  __syncthreads();
  for (int i = tid; i < 2 * T_TOK; i += 256) atomicAdd(&hist[routee[i]], 1);
  __syncthreads();
  if (tid == 0) {
    int off = 0;
    for (int e = 0; e < NEXP; e++) {
      pb[e] = off;
      meta[e] = hist[e];
      off += ((hist[e] + TILE_M - 1) / TILE_M) * TILE_M;
    }
    pb[NEXP] = off;
    ntil = off / TILE_M;
    for (int e = 0; e <= NEXP; e++) meta[8 + e] = pb[e];
    meta[17] = ntil;
  }
  __syncthreads();
  for (int tI = tid; tI < ntil; tI += 256) {
    int r = tI * TILE_M;
    int e = 0;
    while (!(r < pb[e + 1])) e++;
    meta[18 + tI] = e;
  }
  // zero the pad-token row (token id == T_TOK gathers zeros)
  for (int c = tid; c < DIM; c += 256) tok[(size_t)T_TOK * DIM + c] = (bf16_t)0.f;
}

// ---------------------------------------------------------------------------
// 3b) Deterministic placement (wave32 ballot prefix), fills row_token + rowpos
// ---------------------------------------------------------------------------
__global__ void k_place(const int* __restrict__ routee, const int* __restrict__ meta,
                        int* __restrict__ row_token, int* __restrict__ rowpos) {
  int e = blockIdx.x;
  int tid = threadIdx.x;
  int lane = tid & 31, wv = tid >> 5;
  int base = meta[8 + e];
  int cnt = meta[e];
  __shared__ int wsum[8];
  __shared__ int running;
  if (tid == 0) running = 0;
  __syncthreads();
  for (int c0 = 0; c0 < 2 * T_TOK; c0 += 256) {
    int idx = c0 + tid;
    bool match = (routee[idx] == e);
    unsigned mask = __builtin_amdgcn_ballot_w32(match);
    int lrank = __popc(mask & ((1u << lane) - 1u));
    if (lane == 0) wsum[wv] = __popc(mask);
    __syncthreads();
    int pre = 0, tot = 0;
#pragma unroll
    for (int j = 0; j < 8; j++) {
      if (j < wv) pre += wsum[j];
      tot += wsum[j];
    }
    if (match) {
      int pos = base + running + pre + lrank;
      row_token[pos] = idx >> 1;
      rowpos[idx] = pos;
    }
    __syncthreads();
    if (tid == 0) running += tot;
  }
  __syncthreads();
  int padcnt = ((cnt + TILE_M - 1) / TILE_M) * TILE_M;
  for (int p = cnt + tid; p < padcnt; p += 256)
    row_token[base + p] = T_TOK;   // gathers the zero row
}

// ---------------------------------------------------------------------------
// 4) Expert GEMM layer. Block = 32 expert rows x 256 cols; 8 waves each own a
//    32Mx32N macro-tile: per k-step 2 A frags (LDS) + 2 B frags (global/L2)
//    feed 4 WMMAs -> 2x fragment reuse, 4 independent accumulation chains.
// ---------------------------------------------------------------------------
__device__ __forceinline__ v16bf frag16(const bf16_t* p, int hi) {
  bf16x8 lo = *(const bf16x8*)(p + 8 * hi);
  bf16x8 hh = *(const bf16x8*)(p + 16 + 8 * hi);
  return __builtin_shufflevector(lo, hh, 0, 1, 2, 3, 4, 5, 6, 7, 8, 9, 10, 11, 12, 13, 14, 15);
}

template <bool GATHER, bool RELU, bool OUTF32>
__global__ void k_gemm(const bf16_t* __restrict__ Asrc, const int* __restrict__ row_token,
                       const bf16_t* __restrict__ Wt, const int* __restrict__ meta,
                       bf16_t* __restrict__ Obf, float* __restrict__ Of) {
  int bx = blockIdx.x;
  if (bx >= meta[17]) return;               // beyond live tile count
  int e = meta[18 + bx];
  int tid = threadIdx.x;
  int lane = tid & 31, wv = tid >> 5;
  int hi = lane >> 4, nl = lane & 15;
  int rowbase = bx * TILE_M;

  __shared__ bf16_t As[TILE_M * LDS_STRIDE];
  for (int c = tid; c < TILE_M * 64; c += 256) {  // rows x 64 chunks of 8 bf16
    int r = c >> 6, kc = (c & 63) << 3;
    const bf16_t* src = GATHER
        ? Asrc + (size_t)row_token[rowbase + r] * DIM + kc
        : Asrc + (size_t)(rowbase + r) * DIM + kc;
    *(bf16x8*)(&As[r * LDS_STRIDE + kc]) = *(const bf16x8*)src;
  }
  __syncthreads();

  int nb = blockIdx.y * 256 + wv * 32;
  const bf16_t* B0 = Wt + ((size_t)e * DIM + nb + nl) * DIM;
  const bf16_t* B1 = B0 + (size_t)16 * DIM;
  const bf16_t* A0 = &As[nl * LDS_STRIDE];
  const bf16_t* A1 = &As[(16 + nl) * LDS_STRIDE];

  v8f acc[2][2] = {};
#pragma unroll 2
  for (int k0 = 0; k0 < DIM; k0 += 32) {
    v16bf a0 = frag16(A0 + k0, hi);
    v16bf a1 = frag16(A1 + k0, hi);
    v16bf b0 = frag16(B0 + k0, hi);
    v16bf b1 = frag16(B1 + k0, hi);
    acc[0][0] = __builtin_amdgcn_wmma_f32_16x16x32_bf16(false, a0, false, b0, (short)0,
                                                        acc[0][0], false, false);
    acc[1][0] = __builtin_amdgcn_wmma_f32_16x16x32_bf16(false, a1, false, b0, (short)0,
                                                        acc[1][0], false, false);
    acc[0][1] = __builtin_amdgcn_wmma_f32_16x16x32_bf16(false, a0, false, b1, (short)0,
                                                        acc[0][1], false, false);
    acc[1][1] = __builtin_amdgcn_wmma_f32_16x16x32_bf16(false, a1, false, b1, (short)0,
                                                        acc[1][1], false, false);
  }

#pragma unroll
  for (int mi = 0; mi < 2; mi++) {
#pragma unroll
    for (int ni = 0; ni < 2; ni++) {
#pragma unroll
      for (int r = 0; r < 8; r++) {         // C/D: row = r + 8*hi, col = nl
        float v = acc[mi][ni][r];
        if (RELU) v = v > 0.f ? v : 0.f;
        size_t oidx = (size_t)(rowbase + mi * 16 + r + 8 * hi) * DIM + nb + ni * 16 + nl;
        if (OUTF32) Of[oidx] = v;
        else Obf[oidx] = (bf16_t)v;
      }
    }
  }
}

// ---------------------------------------------------------------------------
// 5) Combine: out[b][f][h][w] = w0*Y[pos0][f] + w1*Y[pos1][f]; coalesced on w.
// ---------------------------------------------------------------------------
__global__ void k_combine(const float* __restrict__ Y, const int* __restrict__ rowpos,
                          const float* __restrict__ routew, float* __restrict__ out) {
  int blk = blockIdx.x;            // (b*64 + h)*128 + fgroup
  int fg = blk & 127, bh = blk >> 7;
  int b = bh >> 6, h = bh & 63;
  int f = fg * 4 + (threadIdx.x >> 6);
  int w = threadIdx.x & 63;
  int t = (b << 12) + (h << 6) + w;
  int r0 = rowpos[2 * t], r1 = rowpos[2 * t + 1];
  float v = routew[2 * t] * Y[(size_t)r0 * DIM + f] +
            routew[2 * t + 1] * Y[(size_t)r1 * DIM + f];
  out[(((size_t)(b * DIM + f)) << 12) + (h << 6) + w] = v;
}

// ---------------------------------------------------------------------------
// Launcher
// ---------------------------------------------------------------------------
static inline size_t alignup(size_t x) { return (x + 255) & ~(size_t)255; }

extern "C" void kernel_launch(void* const* d_in, const int* in_sizes, int n_in,
                              void* d_out, int out_size, void* d_ws, size_t ws_size,
                              hipStream_t stream) {
  (void)in_sizes; (void)n_in; (void)out_size; (void)ws_size;
  const float* hs   = (const float*)d_in[0];
  const float* gate = (const float*)d_in[1];
  const float* w1   = (const float*)d_in[2];
  const float* w2   = (const float*)d_in[3];
  const float* w3   = (const float*)d_in[4];
  float* out = (float*)d_out;

  char* ws = (char*)d_ws;
  size_t off = 0;
  bf16_t* tok = (bf16_t*)(ws + off); off = alignup(off + (size_t)(T_TOK + 1) * DIM * 2);
  bf16_t* w1t = (bf16_t*)(ws + off); off = alignup(off + (size_t)NEXP * DIM * DIM * 2);
  bf16_t* w2t = (bf16_t*)(ws + off); off = alignup(off + (size_t)NEXP * DIM * DIM * 2);
  bf16_t* w3t = (bf16_t*)(ws + off); off = alignup(off + (size_t)NEXP * DIM * DIM * 2);
  float*  Y   = (float*)(ws + off);          // 67.6 MB; H1 (bf16) aliases its start
  bf16_t* H1  = (bf16_t*)Y;                  // lifetime: layers 1-2 only
  off = alignup(off + (size_t)P_CAP * DIM * 4);
  bf16_t* H2  = (bf16_t*)(ws + off); off = alignup(off + (size_t)P_CAP * DIM * 2);
  int*   routee = (int*)(ws + off);   off = alignup(off + (size_t)2 * T_TOK * 4);
  float* routew = (float*)(ws + off); off = alignup(off + (size_t)2 * T_TOK * 4);
  int*   row_token = (int*)(ws + off); off = alignup(off + (size_t)P_CAP * 4);
  int*   rowpos = (int*)(ws + off);    off = alignup(off + (size_t)2 * T_TOK * 4);
  int*   meta = (int*)(ws + off);      off = alignup(off + (size_t)(18 + MAX_TILES) * 4);

  k_router<<<dim3(256), dim3(256), 0, stream>>>(hs, gate, tok, routee, routew);
  k_convw<<<dim3(2048), dim3(256), 0, stream>>>(w1, w1t);
  k_convw<<<dim3(2048), dim3(256), 0, stream>>>(w2, w2t);
  k_convw<<<dim3(2048), dim3(256), 0, stream>>>(w3, w3t);
  k_count<<<dim3(1), dim3(256), 0, stream>>>(routee, meta, tok);
  k_place<<<dim3(NEXP), dim3(256), 0, stream>>>(routee, meta, row_token, rowpos);

  dim3 gg(MAX_TILES, 2);
  k_gemm<true,  true,  false><<<gg, dim3(256), 0, stream>>>(tok, row_token, w1t, meta, H1, nullptr);
  k_gemm<false, true,  false><<<gg, dim3(256), 0, stream>>>(H1,  row_token, w2t, meta, H2, nullptr);
  k_gemm<false, false, true ><<<gg, dim3(256), 0, stream>>>(H2,  row_token, w3t, meta, nullptr, Y);

  k_combine<<<dim3(4 * 64 * 128), dim3(256), 0, stream>>>(Y, rowpos, routew, out);
}
